// BaseGenerator_55087250539218
// MI455X (gfx1250) — matrix operand order repaired and checked by reading, not compiled, not run on hardware
//
#include <hip/hip_runtime.h>
#include <hip/hip_bf16.h>
#include <math.h>

// ---------------- problem constants (from reference) ----------------
#define L_   4
#define E_   1024
#define H_   16
#define F_   4096
#define B_   8
#define S_   512
#define DH_  64
#define NGEN 1200        // VV*VR
#define M_   (B_ * S_)   // 4096 tokens
#define PAD_ID 0
#define NEG_BIG 1.0e30f

typedef __bf16 bf16;
typedef bf16  v8bf  __attribute__((ext_vector_type(8)));
typedef bf16  v16bf __attribute__((ext_vector_type(16)));
typedef float v8f   __attribute__((ext_vector_type(8)));
typedef unsigned int uint32x4 __attribute__((ext_vector_type(4)));
typedef int int32x8 __attribute__((ext_vector_type(8)));
typedef int int32x4 __attribute__((ext_vector_type(4)));

__device__ inline v16bf mk16(v8bf lo, v8bf hi) {
  v16bf r;
#pragma unroll
  for (int i = 0; i < 8; ++i) { r[i] = lo[i]; r[i + 8] = hi[i]; }
  return r;
}

__device__ inline float rowmax16(float v) {
#pragma unroll
  for (int m = 1; m < 16; m <<= 1) v = fmaxf(v, __shfl_xor(v, m, 16));
  return v;
}
__device__ inline float rowsum16(float v) {
#pragma unroll
  for (int m = 1; m < 16; m <<= 1) v += __shfl_xor(v, m, 16);
  return v;
}

// ---------------- Tensor Data Mover: 2-D strided tile -> LDS with row padding ----
// D# layout per CDNA5 ISA 8.3/8.4 (group0: flags/lds/global/type; group1: sizes).
// pad_iv: log2(DWORDs-between-pads)-1 (3 => 16 DW = 32 bf16); pad_amt: DWORDs-1.
__device__ inline void tdm_load_2d(unsigned lds_addr, const void* gptr,
                                   unsigned tile_cols, unsigned tile_rows,
                                   unsigned tensor_cols, unsigned tensor_rows,
                                   unsigned row_stride_elems,
                                   unsigned pad_iv, unsigned pad_amt) {
  unsigned long long ga = (unsigned long long)(uintptr_t)gptr;
  uint32x4 g0;
  g0[0] = 1u;                                            // count=1, user mode
  g0[1] = lds_addr;                                      // LDS byte address
  g0[2] = (unsigned)ga;                                  // global addr [31:0]
  g0[3] = (unsigned)((ga >> 32) & 0x01FFFFFFu) | (2u << 30);  // addr[56:32] | type=2
  int32x8 g1;
  g1[0] = (int)((1u << 16) |            // data_size = 1 -> 2-byte elements
                (1u << 20) |            // pad_enable
                (pad_iv << 22) |        // pad_interval
                (pad_amt << 25));       // pad_amount
  g1[1] = (int)((tensor_cols & 0xFFFFu) << 16);                       // tensor_dim0[15:0]
  g1[2] = (int)((tensor_cols >> 16) | ((tensor_rows & 0xFFFFu) << 16)); // dim0 hi | dim1 lo
  g1[3] = (int)((tensor_rows >> 16) | (tile_cols << 16));             // dim1 hi | tile_dim0
  g1[4] = (int)(tile_rows & 0xFFFFu);                                 // tile_dim1 (tile_dim2=0)
  g1[5] = (int)row_stride_elems;                                      // tensor_dim0_stride lo
  g1[6] = 0;
  g1[7] = 0;
  int32x4 gz = {0, 0, 0, 0};
#if __clang_major__ >= 23
  int32x8 gz8 = {0, 0, 0, 0, 0, 0, 0, 0};
  __builtin_amdgcn_tensor_load_to_lds(g0, g1, gz, gz, gz8, 0);
#else
  __builtin_amdgcn_tensor_load_to_lds(g0, g1, gz, gz, 0);
#endif
}

__device__ inline unsigned lds_off(const void* p) {
  // Flat->LDS aperture mapping truncates to addr[31:0] (ISA 10.2).
  return (unsigned)(uintptr_t)p;
}

// ---------------- elementwise helpers ----------------
__global__ void cvt_bf16_kernel(const float* __restrict__ in, bf16* __restrict__ out, int n) {
  int i = blockIdx.x * 256 + threadIdx.x;
  if (i < n) out[i] = (bf16)in[i];
}

__global__ void embed_kernel(const float* __restrict__ ve, const float* __restrict__ re,
                             const int* __restrict__ vs, const int* __restrict__ rs,
                             float* __restrict__ x, bf16* __restrict__ xb) {
  int idx = blockIdx.x * 256 + threadIdx.x;
  if (idx >= M_ * E_) return;
  int tok = idx >> 10;
  int e   = idx & (E_ - 1);
  float t = (ve[(size_t)vs[tok] * E_ + e] + re[(size_t)rs[tok] * E_ + e]) * 32.0f; // sqrt(E)
  x[idx]  = t;
  xb[idx] = (bf16)t;
}

// x_new = LN(x + y); writes fp32 residual stream + bf16 copy (next GEMM input).
__global__ void resln_kernel(const float* __restrict__ xin, const float* __restrict__ yin,
                             const float* __restrict__ g, const float* __restrict__ be,
                             float* __restrict__ xout, bf16* __restrict__ xbout) {
  __shared__ float red[2][8];
  const int row  = blockIdx.x;
  const int tid  = threadIdx.x;
  const int lane = tid & 31, wave = tid >> 5;
  float v[4];
  float s = 0.f, ss = 0.f;
#pragma unroll
  for (int i = 0; i < 4; ++i) {
    int e = tid + i * 256;
    float t = xin[(size_t)row * E_ + e];
    if (yin) t += yin[(size_t)row * E_ + e];
    v[i] = t; s += t; ss += t * t;
  }
#pragma unroll
  for (int m = 1; m < 32; m <<= 1) { s += __shfl_xor(s, m, 32); ss += __shfl_xor(ss, m, 32); }
  if (lane == 0) { red[0][wave] = s; red[1][wave] = ss; }
  __syncthreads();
  s = 0.f; ss = 0.f;
#pragma unroll
  for (int w = 0; w < 8; ++w) { s += red[0][w]; ss += red[1][w]; }
  float mean = s * (1.0f / E_);
  float var  = ss * (1.0f / E_) - mean * mean;
  float rstd = rsqrtf(var + 1e-5f);
#pragma unroll
  for (int i = 0; i < 4; ++i) {
    int e = tid + i * 256;
    float t = (v[i] - mean) * rstd * g[e] + be[e];
    xout[(size_t)row * E_ + e]  = t;
    xbout[(size_t)row * E_ + e] = (bf16)t;
  }
}

// ---------------- tiled bf16 WMMA GEMM: C[M,N] = A[M,K] * W[N,K]^T + bias ----------------
// TDM double-buffered staging: wave 0 drives the Tensor Data Mover, compute waves
// only do ds_load_b128 fragment reads + v_wmma. LDS rows padded to 40 elems by TDM.
enum { EPI_BF16 = 0, EPI_BF16_GELU = 1, EPI_F32 = 2 };

template <int EPI>
__global__ __launch_bounds__(256)
void gemm_bf16_wmma(const bf16* __restrict__ A, const bf16* __restrict__ W,
                    const float* __restrict__ bias,
                    float* __restrict__ Cf, bf16* __restrict__ Cb,
                    int M, int N, int K) {
  __shared__ bf16 sA[2][128][40];
  __shared__ bf16 sB[2][128][40];
  const int tid  = threadIdx.x;
  const int lane = tid & 31;
  const int wave = tid >> 5;
  const int m0 = blockIdx.y * 128;
  const int n0 = blockIdx.x * 128;
  const int wm = (wave & 3) * 32;   // 4 waves along M
  const int wn = (wave >> 2) * 64;  // 2 waves along N
  const int rsel  = lane & 15;
  const int akofs = (lane < 16) ? 0 : 8;    // 16-bit A-matrix layout (ISA 7.12.2)
  const int bkofs = (lane < 16) ? 0 : 8;    // 16-bit B-matrix layout
  const int rofs  = (lane < 16) ? 0 : 8;    // C/D layout

  v8f acc[2][4];
#pragma unroll
  for (int mf = 0; mf < 2; ++mf)
#pragma unroll
    for (int nf = 0; nf < 4; ++nf)
#pragma unroll
      for (int i = 0; i < 8; ++i) acc[mf][nf][i] = 0.f;

  const int nt = K / 32;
  if (wave == 0) {   // prologue: tile 0 into buffer 0
    tdm_load_2d(lds_off(&sA[0][0][0]), &A[(size_t)m0 * K], 32, 128,
                (unsigned)K, (unsigned)(M - m0), (unsigned)K, 3, 3);
    tdm_load_2d(lds_off(&sB[0][0][0]), &W[(size_t)n0 * K], 32, 128,
                (unsigned)K, (unsigned)(N - n0), (unsigned)K, 3, 3);
  }

  for (int t = 0; t < nt; ++t) {
    const int buf = t & 1;
    if (wave == 0) {
      if (t + 1 < nt) {   // prefetch next tile into the other buffer (TDM in-order)
        const int k1 = (t + 1) * 32;
        tdm_load_2d(lds_off(&sA[buf ^ 1][0][0]), &A[(size_t)m0 * K + k1], 32, 128,
                    (unsigned)(K - k1), (unsigned)(M - m0), (unsigned)K, 3, 3);
        tdm_load_2d(lds_off(&sB[buf ^ 1][0][0]), &W[(size_t)n0 * K + k1], 32, 128,
                    (unsigned)(K - k1), (unsigned)(N - n0), (unsigned)K, 3, 3);
        __builtin_amdgcn_s_wait_tensorcnt((short)2);  // drain current tile's 2 ops
      } else {
        __builtin_amdgcn_s_wait_tensorcnt((short)0);
      }
    }
    __syncthreads();

    v16bf af[2], bfr[4];
#pragma unroll
    for (int mf = 0; mf < 2; ++mf) {
      int r = wm + mf * 16 + rsel;
      af[mf] = mk16(*(const v8bf*)&sA[buf][r][akofs], *(const v8bf*)&sA[buf][r][akofs + 16]);
    }
#pragma unroll
    for (int nf = 0; nf < 4; ++nf) {
      int c = wn + nf * 16 + rsel;
      bfr[nf] = mk16(*(const v8bf*)&sB[buf][c][bkofs], *(const v8bf*)&sB[buf][c][bkofs + 16]);
    }
#pragma unroll
    for (int mf = 0; mf < 2; ++mf)
#pragma unroll
      for (int nf = 0; nf < 4; ++nf)
        acc[mf][nf] = __builtin_amdgcn_wmma_f32_16x16x32_bf16(
            false, af[mf], false, bfr[nf], (short)0, acc[mf][nf], false, false);
    __syncthreads();
  }

#pragma unroll
  for (int mf = 0; mf < 2; ++mf)
#pragma unroll
    for (int nf = 0; nf < 4; ++nf) {
      int n = n0 + wn + nf * 16 + rsel;
      if (n >= N) continue;
      float bval = bias ? bias[n] : 0.f;
#pragma unroll
      for (int vr = 0; vr < 8; ++vr) {
        int m = m0 + wm + mf * 16 + rofs + vr;
        float v = acc[mf][nf][vr] + bval;
        if (EPI == EPI_BF16_GELU) v = 0.5f * v * (1.f + erff(v * 0.70710678118f));
        if (EPI == EPI_F32) Cf[(size_t)m * N + n] = v;
        else                Cb[(size_t)m * N + n] = (bf16)v;
      }
    }
}

// ---------------- flash attention (causal + pad + distance bias) ----------------
// grid: (S/128, H, B); block 256 = 8 waves; each wave owns 16 query rows.
// K tile staged by TDM (row stride 3E, padded to 72); V tile staged TRANSPOSED so the
// P@V B-fragment read is two ds_load_b128 instead of 16 scalar ds_load_u16.
__global__ __launch_bounds__(256)
void attention_wmma(const bf16* __restrict__ qkv, const int* __restrict__ valseq,
                    const int* __restrict__ dist, const float* __restrict__ dist_emb,
                    bf16* __restrict__ ctx) {
  __shared__ bf16 sK[32][72];     // [key][d], TDM-padded rows
  __shared__ bf16 sVt[64][40];    // [d][key], transposed at staging
  __shared__ bf16 sP[8][16][32];  // per-wave P transpose buffer
  const int tid  = threadIdx.x;
  const int lane = tid & 31;
  const int wave = tid >> 5;
  const int b = blockIdx.z, h = blockIdx.y;
  const int qbase = blockIdx.x * 128 + wave * 16;
  const int rsel  = lane & 15;
  const int akofs = (lane < 16) ? 0 : 8;
  const int bkofs = (lane < 16) ? 0 : 8;
  const int rofs  = (lane < 16) ? 0 : 8;
  const size_t strideS = (size_t)3 * E_;
  const bf16* qp = qkv + (size_t)(b * S_) * strideS + h * DH_;
  const bf16* kp = qp + E_;
  const bf16* vp = qp + 2 * E_;

  // Q fragments (A layout), dh=64 -> two 32-wide inner chunks
  v16bf qf[2];
  {
    const bf16* qq = qp + (size_t)(qbase + rsel) * strideS;
#pragma unroll
    for (int c = 0; c < 2; ++c)
      qf[c] = mk16(*(const v8bf*)&qq[32 * c + akofs], *(const v8bf*)&qq[32 * c + akofs + 16]);
  }

  v8f acco[4];
#pragma unroll
  for (int nf = 0; nf < 4; ++nf)
#pragma unroll
    for (int i = 0; i < 8; ++i) acco[nf][i] = 0.f;
  float rmax[8], rsum[8];
#pragma unroll
  for (int i = 0; i < 8; ++i) { rmax[i] = -NEG_BIG; rsum[i] = 0.f; }

  const int nkt = (blockIdx.x * 128 + 127) / 32 + 1;   // block-max causal extent
  for (int kt = 0; kt < nkt; ++kt) {
    const int kb = kt * 32;
    __syncthreads();   // previous tile fully consumed
    if (wave == 0) {   // K tile via TDM: 64 cols x 32 rows, pad 32DW rows by 4DW -> stride 72
      tdm_load_2d(lds_off(&sK[0][0]), &kp[(size_t)kb * strideS], 64, 32,
                  64u, (unsigned)(S_ - kb), (unsigned)strideS, 4, 3);
    }
    {   // V tile staged transposed: each thread loads 8 contiguous dims of one key
      int krow = tid >> 3;
      int dc   = (tid & 7) * 8;
      const bf16* vg = &vp[(size_t)(kb + krow) * strideS + dc];
      v8bf vv = *(const v8bf*)vg;
      if (kb + 32 + krow < S_) __builtin_prefetch(vg + 32 * strideS, 0, 1);
#pragma unroll
      for (int i = 0; i < 8; ++i) sVt[dc + i][krow] = vv[i];
    }
    if (wave == 0) __builtin_amdgcn_s_wait_tensorcnt((short)0);
    __syncthreads();
    if (kb > qbase) continue;   // wave-uniform causal skip (still joins barriers)

    // scores: two 16-key column tiles, inner = dh via 2x wmma each
    v8f st[2];
#pragma unroll
    for (int j = 0; j < 2; ++j) {
#pragma unroll
      for (int i = 0; i < 8; ++i) st[j][i] = 0.f;
      const int key = 16 * j + rsel;
#pragma unroll
      for (int c = 0; c < 2; ++c) {
        v16bf kf = mk16(*(const v8bf*)&sK[key][32 * c + bkofs],
                        *(const v8bf*)&sK[key][32 * c + bkofs + 16]);
        st[j] = __builtin_amdgcn_wmma_f32_16x16x32_bf16(false, qf[c], false, kf,
                                                        (short)0, st[j], false, false);
      }
      const int kg = kb + key;
      const bool kpad = (valseq[b * S_ + kg] == PAD_ID);
#pragma unroll
      for (int vr = 0; vr < 8; ++vr) {
        int q = qbase + rofs + vr;
        float sv = st[j][vr] * 0.125f;   // 1/sqrt(dh)
        if (kg > q || kpad) sv = -NEG_BIG;
        else sv += dist_emb[(size_t)dist[((size_t)(b * S_ + q)) * S_ + kg] * H_ + h];
        st[j][vr] = sv;
      }
    }

    // online softmax update (row stats live in the 16-lane group matching C layout)
#pragma unroll
    for (int vr = 0; vr < 8; ++vr) {
      float tm = rowmax16(fmaxf(st[0][vr], st[1][vr]));
      float nm = fmaxf(rmax[vr], tm);
      float scale = expf(rmax[vr] - nm);
      float e0 = expf(st[0][vr] - nm);
      float e1 = expf(st[1][vr] - nm);
      rsum[vr] = rsum[vr] * scale + rowsum16(e0 + e1);
      rmax[vr] = nm;
      st[0][vr] = e0; st[1][vr] = e1;
#pragma unroll
      for (int nf = 0; nf < 4; ++nf) acco[nf][vr] *= scale;
    }

    // transpose P (C layout -> A layout) through per-wave LDS
#pragma unroll
    for (int j = 0; j < 2; ++j)
#pragma unroll
      for (int vr = 0; vr < 8; ++vr)
        sP[wave][rofs + vr][16 * j + rsel] = (bf16)st[j][vr];
    asm volatile("s_wait_dscnt 0" ::: "memory");
    v16bf pf = mk16(*(const v8bf*)&sP[wave][rsel][akofs],
                    *(const v8bf*)&sP[wave][rsel][akofs + 16]);

    // O += P @ V   (inner = 32 keys; B-frag now contiguous from transposed V)
#pragma unroll
    for (int nf = 0; nf < 4; ++nf) {
      const int d = nf * 16 + rsel;
      v16bf vf = mk16(*(const v8bf*)&sVt[d][bkofs], *(const v8bf*)&sVt[d][bkofs + 16]);
      acco[nf] = __builtin_amdgcn_wmma_f32_16x16x32_bf16(false, pf, false, vf,
                                                         (short)0, acco[nf], false, false);
    }
  }

  // epilogue: ctx[b, q, h*64 + d] = O / rowsum
#pragma unroll
  for (int nf = 0; nf < 4; ++nf) {
    const int d = nf * 16 + rsel;
#pragma unroll
    for (int vr = 0; vr < 8; ++vr) {
      int q = qbase + rofs + vr;
      float o = acco[nf][vr] / rsum[vr];
      ctx[((size_t)(b * S_ + q)) * E_ + h * DH_ + d] = (bf16)o;
    }
  }
}

// ---------------- host orchestration ----------------
static inline int ceil_div(int a, int b) { return (a + b - 1) / b; }

extern "C" void kernel_launch(void* const* d_in, const int* in_sizes, int n_in,
                              void* d_out, int out_size, void* d_ws, size_t ws_size,
                              hipStream_t stream) {
  const float* val_emb  = (const float*)d_in[0];
  const float* ring_emb = (const float*)d_in[1];
  const float* dist_emb = (const float*)d_in[2];
  const float* Wqkv = (const float*)d_in[3];
  const float* bqkv = (const float*)d_in[4];
  const float* Wo   = (const float*)d_in[5];
  const float* bo   = (const float*)d_in[6];
  const float* ln1s = (const float*)d_in[7];
  const float* ln1b = (const float*)d_in[8];
  const float* W1   = (const float*)d_in[9];
  const float* b1   = (const float*)d_in[10];
  const float* W2   = (const float*)d_in[11];
  const float* b2   = (const float*)d_in[12];
  const float* ln2s = (const float*)d_in[13];
  const float* ln2b = (const float*)d_in[14];
  const float* lnfs = (const float*)d_in[15];
  const float* lnfb = (const float*)d_in[16];
  const float* genW = (const float*)d_in[17];
  const float* genb = (const float*)d_in[18];
  const int* vs   = (const int*)d_in[19];
  const int* rs   = (const int*)d_in[20];
  const int* dist = (const int*)d_in[21];
  float* out = (float*)d_out;

  // workspace carve (256B aligned)
  size_t off = 0;
  char* base = (char*)d_ws;
  auto carve = [&](size_t bytes) -> void* {
    void* p = base + off;
    off += (bytes + 255) & ~(size_t)255;
    return p;
  };
  bf16* wqkv_b = (bf16*)carve((size_t)L_ * 3 * E_ * E_ * 2);
  bf16* wo_b   = (bf16*)carve((size_t)L_ * E_ * E_ * 2);
  bf16* w1_b   = (bf16*)carve((size_t)L_ * F_ * E_ * 2);
  bf16* w2_b   = (bf16*)carve((size_t)L_ * E_ * F_ * 2);
  bf16* gw_b   = (bf16*)carve((size_t)NGEN * E_ * 2);
  float* x     = (float*)carve((size_t)M_ * E_ * 4);
  bf16*  xb    = (bf16*) carve((size_t)M_ * E_ * 2);
  bf16*  qkvb  = (bf16*) carve((size_t)M_ * 3 * E_ * 2);
  bf16*  ctxb  = (bf16*) carve((size_t)M_ * E_ * 2);
  bf16*  ffhb  = (bf16*) carve((size_t)M_ * F_ * 2);
  float* tmp   = (float*)carve((size_t)M_ * E_ * 4);

  auto cvt = [&](const float* src, bf16* dst, int n) {
    cvt_bf16_kernel<<<ceil_div(n, 256), 256, 0, stream>>>(src, dst, n);
  };
  cvt(Wqkv, wqkv_b, L_ * 3 * E_ * E_);
  cvt(Wo,   wo_b,   L_ * E_ * E_);
  cvt(W1,   w1_b,   L_ * F_ * E_);
  cvt(W2,   w2_b,   L_ * E_ * F_);
  cvt(genW, gw_b,   NGEN * E_);

  embed_kernel<<<ceil_div(M_ * E_, 256), 256, 0, stream>>>(val_emb, ring_emb, vs, rs, x, xb);

  for (int l = 0; l < L_; ++l) {
    // QKV projection -> bf16 [M, 3E]
    gemm_bf16_wmma<EPI_BF16><<<dim3(3 * E_ / 128, M_ / 128), 256, 0, stream>>>(
        xb, wqkv_b + (size_t)l * 3 * E_ * E_, bqkv + (size_t)l * 3 * E_,
        nullptr, qkvb, M_, 3 * E_, E_);
    // flash attention with on-the-fly distance-bias gather
    attention_wmma<<<dim3(S_ / 128, H_, B_), 256, 0, stream>>>(qkvb, vs, dist, dist_emb, ctxb);
    // output projection -> fp32, then residual + LN1
    gemm_bf16_wmma<EPI_F32><<<dim3(E_ / 128, M_ / 128), 256, 0, stream>>>(
        ctxb, wo_b + (size_t)l * E_ * E_, bo + (size_t)l * E_, tmp, nullptr, M_, E_, E_);
    resln_kernel<<<M_, 256, 0, stream>>>(x, tmp, ln1s + (size_t)l * E_, ln1b + (size_t)l * E_, x, xb);
    // FFN: GEMM+GELU (bf16 out), GEMM (fp32 out), residual + LN2
    gemm_bf16_wmma<EPI_BF16_GELU><<<dim3(F_ / 128, M_ / 128), 256, 0, stream>>>(
        xb, w1_b + (size_t)l * F_ * E_, b1 + (size_t)l * F_, nullptr, ffhb, M_, F_, E_);
    gemm_bf16_wmma<EPI_F32><<<dim3(E_ / 128, M_ / 128), 256, 0, stream>>>(
        ffhb, w2_b + (size_t)l * E_ * F_, b2 + (size_t)l * E_, tmp, nullptr, M_, E_, F_);
    resln_kernel<<<M_, 256, 0, stream>>>(x, tmp, ln2s + (size_t)l * E_, ln2b + (size_t)l * E_, x, xb);
  }

  // final LN then generator head -> fp32 logits [M, 1200] (N guarded by TDM OOB zero-fill)
  resln_kernel<<<M_, 256, 0, stream>>>(x, nullptr, lnfs, lnfb, x, xb);
  gemm_bf16_wmma<EPI_F32><<<dim3(ceil_div(NGEN, 128), M_ / 128), 256, 0, stream>>>(
      xb, gw_b, genb, out, nullptr, M_, NGEN, E_);
}
